// TokenPose_PPT_base_61040075210879
// MI455X (gfx1250) — matrix operand order repaired and checked by reading, not compiled, not run on hardware
//
#include <hip/hip_runtime.h>
#include <math.h>

typedef __attribute__((ext_vector_type(16))) __bf16 v16bf;
typedef __attribute__((ext_vector_type(8)))  __bf16 v8bf;
typedef __attribute__((ext_vector_type(8)))  float  v8f;

static constexpr int Bn      = 32;
static constexpr int Hn      = 12;
static constexpr int Dm      = 192;   // DIM
static constexpr int Jn      = 17;    // joint tokens
static constexpr int P0n     = 256;   // initial patches
static constexpr int MLPn    = 576;
static constexpr int NMAX    = 273;   // J + P0
static constexpr int NMAXPAD = 288;   // padded to 16

// ---------------------------------------------------------------------------
// Weight prep: fp32 [L][K][N] -> bf16 transposed [L][N][K]
// ---------------------------------------------------------------------------
__global__ void wt_prep(const float* __restrict__ W, __bf16* __restrict__ Wt,
                        int L, int K, int N)
{
    size_t t = (size_t)blockIdx.x * blockDim.x + threadIdx.x;
    size_t total = (size_t)L * K * N;
    if (t >= total) return;
    int nn = (int)(t % N);
    int kk = (int)((t / N) % K);
    int l  = (int)(t / ((size_t)N * K));
    Wt[((size_t)l * N + nn) * K + kk] = (__bf16)W[t];
}

__global__ void init_pos(const float* __restrict__ src, float* __restrict__ pos)
{
    int t = blockIdx.x * blockDim.x + threadIdx.x;
    if (t >= Bn * P0n * Dm) return;
    pos[t] = src[t % (P0n * Dm)];
}

__global__ void zero_f(float* __restrict__ p, int count)
{
    int t = blockIdx.x * blockDim.x + threadIdx.x;
    if (t < count) p[t] = 0.0f;
}

// x[:, J:] += pos * la[l];  pos *= la[l]   (x packed [B][n][D], pos [B][P][D])
__global__ void add_pos(float* __restrict__ x, float* __restrict__ pos,
                        const float* __restrict__ la, int l, int P, int n)
{
    int t = blockIdx.x * blockDim.x + threadIdx.x;
    int total = Bn * P * Dm;
    if (t >= total) return;
    int c = t % Dm;
    int p = (t / Dm) % P;
    int b = t / (Dm * P);
    float s  = la[l];
    float pv = pos[t] * s;
    pos[t] = pv;
    x[((size_t)b * n + Jn + p) * Dm + c] += pv;
}

// ---------------------------------------------------------------------------
// LayerNorm over D=192, one wave per row, output bf16
// ---------------------------------------------------------------------------
__global__ __launch_bounds__(256)
void ln_rows(const float* __restrict__ x, const float* __restrict__ w,
             const float* __restrict__ b, __bf16* __restrict__ out, int M)
{
    int row  = blockIdx.x * 8 + (threadIdx.x >> 5);
    int lane = threadIdx.x & 31;
    if (row >= M) return;
    const float* xr = x + (size_t)row * Dm;
    float s = 0.f, ss = 0.f;
    for (int i = lane; i < Dm; i += 32) { float v = xr[i]; s += v; ss += v * v; }
    for (int o = 16; o > 0; o >>= 1) { s += __shfl_xor(s, o, 32); ss += __shfl_xor(ss, o, 32); }
    float mean = s * (1.0f / Dm);
    float var  = ss * (1.0f / Dm) - mean * mean;
    float inv  = rsqrtf(var + 1e-5f);
    __bf16* orow = out + (size_t)row * Dm;
    for (int i = lane; i < Dm; i += 32)
        orow[i] = (__bf16)((xr[i] - mean) * inv * w[i] + b[i]);
}

// ---------------------------------------------------------------------------
// GEMM: C[M][N] = A[M][K](bf16) @ Bt[N][K](bf16)^T, fp32 accumulate via WMMA.
// Each wave computes a 16x64 strip (4 accumulators): A fragment is loaded
// once per K-step and reused across 4 B tiles -> 2.5 loads per v_wmma.
// MODE 0: store f32. MODE 1: +bias +residual, store f32 (in-place safe).
// MODE 2: +bias, exact GELU, store bf16.
// ---------------------------------------------------------------------------
template <int MODE>
__global__ __launch_bounds__(128)
void gemm_bf16_wmma(const __bf16* __restrict__ A, const __bf16* __restrict__ Bt,
                    const float* __restrict__ bias, const float* __restrict__ resid,
                    float* __restrict__ Cf, __bf16* __restrict__ Cb,
                    int M, int N, int K)
{
    const int lane = threadIdx.x & 31;
    const int wave = threadIdx.x >> 5;
    const int m0 = (blockIdx.x * 4 + wave) * 16;
    const int n0 = blockIdx.y * 64;
    if (m0 >= M) return;                      // wave-uniform

    const int r   = lane & 15;
    const int hiA = (lane >> 4) * 8;          // A-fragment K offset
    const int hiB = (lane >> 4) * 8;          // B-fragment K offset

    const __bf16* Ap  = A  + (size_t)(m0 + r) * K;
    const __bf16* Bp0 = Bt + (size_t)(n0 + r) * K + hiB;

    v8f acc[4] = {};
    for (int kt = 0; kt < K; kt += 32) {
        v8bf a0 = *(const v8bf*)(Ap + kt + hiA);
        v8bf a1 = *(const v8bf*)(Ap + kt + 16 + hiA);
        v16bf a;
#pragma unroll
        for (int i = 0; i < 8; ++i) { a[i] = a0[i]; a[i + 8] = a1[i]; }
#pragma unroll
        for (int nt = 0; nt < 4; ++nt) {
            const __bf16* Bp = Bp0 + (size_t)nt * 16 * K + kt;
            v8bf b0 = *(const v8bf*)(Bp);
            v8bf b1 = *(const v8bf*)(Bp + 16);
            v16bf b;
#pragma unroll
            for (int i = 0; i < 8; ++i) { b[i] = b0[i]; b[i + 8] = b1[i]; }
            acc[nt] = __builtin_amdgcn_wmma_f32_16x16x32_bf16(false, a, false, b,
                                                              (short)0, acc[nt], false, false);
        }
    }

    const int rof = (lane >> 4) * 8;
#pragma unroll
    for (int nt = 0; nt < 4; ++nt) {
        const int col = n0 + nt * 16 + (lane & 15);
#pragma unroll
        for (int e = 0; e < 8; ++e) {
            int row = m0 + rof + e;
            float v = acc[nt][e];
            if (MODE >= 1) v += bias[col];
            if (MODE == 1) v += resid[(size_t)row * N + col];
            if (MODE == 2) {
                v = 0.5f * v * (1.0f + erff(v * 0.70710678118654752f));
                Cb[(size_t)row * N + col] = (__bf16)v;
            } else {
                Cf[(size_t)row * N + col] = v;
            }
        }
    }
}

// ---------------------------------------------------------------------------
// RoPE + repack: qkv[B*n][576] f32 ->
//   q bf16 [B][H][npad][16]  (pre-scaled by dh^-0.5)
//   k bf16 [B][H][npad][32]  (upper 16 zero -> divergence-free B fragments)
//   v bf16 [B][H][16][npad]  (transposed)
// Padded rows zeroed.
// ---------------------------------------------------------------------------
__global__ void rope_pack(const float* __restrict__ qkv, const float* __restrict__ ang,
                          __bf16* __restrict__ qb, __bf16* __restrict__ kb,
                          __bf16* __restrict__ vt, int n, int npad)
{
    int t = blockIdx.x * blockDim.x + threadIdx.x;
    int total = Bn * Hn * npad;
    if (t >= total) return;
    int tok = t % npad;
    int h   = (t / npad) % Hn;
    int b   = t / (npad * Hn);
    size_t bh = (size_t)(b * Hn + h);
    __bf16* qo = qb + (bh * npad + tok) * 16;
    __bf16* ko = kb + (bh * npad + tok) * 32;
    __bf16* vo = vt + bh * 16 * npad + tok;
    if (tok >= n) {
#pragma unroll
        for (int i = 0; i < 16; ++i) {
            qo[i] = (__bf16)0.0f; ko[i] = (__bf16)0.0f; ko[16 + i] = (__bf16)0.0f;
            vo[i * npad] = (__bf16)0.0f;
        }
        return;
    }
    const float* base = qkv + ((size_t)b * n + tok) * (3 * Dm) + h * 16;
    const float scale = 0.25f;   // dh^-0.5, dh=16
#pragma unroll
    for (int f = 0; f < 8; ++f) {
        float a  = ang[tok * 8 + f];
        float cv = cosf(a), sv = sinf(a);
        float q0 = base[2 * f],          q1 = base[2 * f + 1];
        float k0 = base[Dm + 2 * f],     k1 = base[Dm + 2 * f + 1];
        float v0 = base[2 * Dm + 2 * f], v1 = base[2 * Dm + 2 * f + 1];
        qo[2 * f]     = (__bf16)((q0 * cv - q1 * sv) * scale);
        qo[2 * f + 1] = (__bf16)((q0 * sv + q1 * cv) * scale);
        ko[2 * f]      = (__bf16)(k0 * cv - k1 * sv);
        ko[2 * f + 1]  = (__bf16)(k0 * sv + k1 * cv);
        ko[16 + 2 * f]     = (__bf16)0.0f;     // zero pad half (K=16..31)
        ko[16 + 2 * f + 1] = (__bf16)0.0f;
        vo[(2 * f) * npad]     = (__bf16)v0;
        vo[(2 * f + 1) * npad] = (__bf16)v1;
    }
}

// ---------------------------------------------------------------------------
// Attention: one wave per (b, h, 16-query block). S = Q K^T via WMMA (dh=16,
// K zero-padded to 32), exact softmax in LDS, O = P V via WMMA.
// Optionally accumulates pruning stats human[b][p] = sum_{h,i<J} attn/H.
// ---------------------------------------------------------------------------
__global__ __launch_bounds__(32)
void attn_kernel(const __bf16* __restrict__ qb, const __bf16* __restrict__ kb,
                 const __bf16* __restrict__ vt, __bf16* __restrict__ obf,
                 float* __restrict__ human, int n, int npad, int do_stats)
{
    const int qblk = blockIdx.x, h = blockIdx.y, b = blockIdx.z;
    const int lane = threadIdx.x;
    __shared__ float  sS[16 * NMAXPAD];
    __shared__ __bf16 sP[16 * NMAXPAD];

    const int r    = lane & 15;
    const int hiA  = (lane >> 4) * 8;
    const int hiB  = (lane >> 4) * 8;
    const size_t bh = (size_t)(b * Hn + h);
    const int q0 = qblk * 16;

    // Q fragment: K = dh = 16 real, upper 16 zero (in-register)
    v16bf aq;
    {
        const __bf16* qp = qb + (bh * npad + q0 + r) * 16 + hiA;
        v8bf q8 = *(const v8bf*)qp;
#pragma unroll
        for (int i = 0; i < 8; ++i) { aq[i] = q8[i]; aq[i + 8] = (__bf16)0.0f; }
    }

    const int nkt = npad >> 4;
    for (int jt = 0; jt < nkt; ++jt) {
        // K rows stored padded to 32: k1 (k=16..31) reads the zero half. No divergence.
        const __bf16* kp = kb + (bh * npad + jt * 16 + r) * 32 + hiB;
        v8bf k0 = *(const v8bf*)kp;
        v8bf k1 = *(const v8bf*)(kp + 16);
        v16bf bk;
#pragma unroll
        for (int i = 0; i < 8; ++i) { bk[i] = k0[i]; bk[i + 8] = k1[i]; }
        v8f s = {};
        s = __builtin_amdgcn_wmma_f32_16x16x32_bf16(false, aq, false, bk,
                                                    (short)0, s, false, false);
        const int rof = (lane >> 4) * 8;
        const int c = jt * 16 + (lane & 15);
#pragma unroll
        for (int e = 0; e < 8; ++e) sS[(rof + e) * NMAXPAD + c] = s[e];
    }
    __syncthreads();

    // Exact softmax per query row (lanes 0-15 each own one row)
    if (lane < 16) {
        float* Sr = sS + lane * NMAXPAD;
        float mx = -3.0e38f;
        for (int j = 0; j < n; ++j) mx = fmaxf(mx, Sr[j]);
        float sum = 0.0f;
        for (int j = 0; j < n; ++j) sum += expf(Sr[j] - mx);
        float inv = 1.0f / sum;
        __bf16* Pr = sP + lane * NMAXPAD;
        for (int j = 0; j < npad; ++j) {
            float p = (j < n) ? expf(Sr[j] - mx) * inv : 0.0f;
            Pr[j] = (__bf16)p;
        }
        if (do_stats && (q0 + lane) < Jn) {
            for (int j = Jn; j < n; ++j) {
                float p = expf(Sr[j] - mx) * inv * (1.0f / (float)Hn);
                atomicAdd(&human[b * P0n + (j - Jn)], p);
            }
        }
    }
    __syncthreads();

    // O = P @ V  (V stored transposed [d][key])
    v8f oacc = {};
    for (int kt = 0; kt < npad; kt += 32) {
        v16bf ap, bv;
        const __bf16* pp = sP + r * NMAXPAD + kt + hiA;
#pragma unroll
        for (int i = 0; i < 8; ++i) { ap[i] = pp[i]; ap[i + 8] = pp[16 + i]; }
        const __bf16* vp = vt + (bh * 16 + r) * npad + kt + hiB;
#pragma unroll
        for (int i = 0; i < 8; ++i) { bv[i] = vp[i]; bv[i + 8] = vp[16 + i]; }
        oacc = __builtin_amdgcn_wmma_f32_16x16x32_bf16(false, ap, false, bv,
                                                       (short)0, oacc, false, false);
    }
    const int rof = (lane >> 4) * 8;
    const int d = lane & 15;
#pragma unroll
    for (int e = 0; e < 8; ++e) {
        int q = q0 + rof + e;
        if (q < n) obf[((size_t)b * n + q) * Dm + h * 16 + d] = (__bf16)oacc[e];
    }
}

// ---------------------------------------------------------------------------
// Top-K per batch (descending values, ties -> lower index, like jax top_k)
// ---------------------------------------------------------------------------
__global__ __launch_bounds__(256)
void topk_kernel(const float* __restrict__ human, int* __restrict__ idx, int P, int Ksel)
{
    __shared__ float sv[256];
    __shared__ float rv[256];
    __shared__ int   ri[256];
    int b = blockIdx.x, t = threadIdx.x;
    sv[t] = (t < P) ? human[b * P0n + t] : -3.0e38f;
    __syncthreads();
    for (int sel = 0; sel < Ksel; ++sel) {
        rv[t] = sv[t]; ri[t] = t;
        __syncthreads();
        for (int s = 128; s > 0; s >>= 1) {
            if (t < s) {
                if (rv[t + s] > rv[t] || (rv[t + s] == rv[t] && ri[t + s] < ri[t])) {
                    rv[t] = rv[t + s]; ri[t] = ri[t + s];
                }
            }
            __syncthreads();
        }
        if (t == 0) { idx[b * P0n + sel] = ri[0]; sv[ri[0]] = -3.0e38f; }
        __syncthreads();
    }
}

// Repack x/pos with selected patches (keeps packed dense layout)
__global__ void gather_kernel(const float* __restrict__ x, const float* __restrict__ pos,
                              const int* __restrict__ idx, float* __restrict__ xo,
                              float* __restrict__ poso, int n, int P, int Ksel)
{
    int nn = Jn + Ksel;
    int t = blockIdx.x * blockDim.x + threadIdx.x;
    int total = Bn * nn * Dm;
    if (t >= total) return;
    int c = t % Dm;
    int r = (t / Dm) % nn;
    int b = t / (Dm * nn);
    int src;
    if (r < Jn) {
        src = r;
    } else {
        int k = idx[b * P0n + (r - Jn)];
        src = Jn + k;
        poso[((size_t)b * Ksel + (r - Jn)) * Dm + c] = pos[((size_t)b * P + k) * Dm + c];
    }
    xo[((size_t)b * nn + r) * Dm + c] = x[((size_t)b * n + src) * Dm + c];
}

// ---------------------------------------------------------------------------
extern "C" void kernel_launch(void* const* d_in, const int* in_sizes, int n_in,
                              void* d_out, int out_size, void* d_ws, size_t ws_size,
                              hipStream_t stream)
{
    (void)in_sizes; (void)n_in; (void)ws_size;
    const float* x_in   = (const float*)d_in[0];
    const float* pos_in = (const float*)d_in[1];
    const float* la     = (const float*)d_in[2];
    const float* ang    = (const float*)d_in[3];
    const float* qkv_w  = (const float*)d_in[4];
    const float* out_w  = (const float*)d_in[5];
    const float* out_b  = (const float*)d_in[6];
    const float* ln1_w  = (const float*)d_in[7];
    const float* ln1_b  = (const float*)d_in[8];
    const float* ff_w1  = (const float*)d_in[9];
    const float* ff_b1  = (const float*)d_in[10];
    const float* ff_w2  = (const float*)d_in[11];
    const float* ff_b2  = (const float*)d_in[12];
    const float* ln2_w  = (const float*)d_in[13];
    const float* ln2_b  = (const float*)d_in[14];

    char* base = (char*)d_ws;
    size_t off = 0;
    auto alloc = [&](size_t bytes) -> char* {
        char* p = base + off;
        off = (off + bytes + 255) & ~(size_t)255;
        return p;
    };

    __bf16* WqkvT = (__bf16*)alloc((size_t)12 * MLPn * Dm * 2);   // [l][576][192]
    __bf16* WoutT = (__bf16*)alloc((size_t)12 * Dm * Dm * 2);     // [l][192][192]
    __bf16* Wff1T = (__bf16*)alloc((size_t)12 * MLPn * Dm * 2);   // [l][576][192]
    __bf16* Wff2T = (__bf16*)alloc((size_t)12 * Dm * MLPn * 2);   // [l][192][576]
    float*  xA    = (float*)alloc((size_t)Bn * NMAX * Dm * 4);
    float*  xB    = (float*)alloc((size_t)Bn * NMAX * Dm * 4);
    float*  posA  = (float*)alloc((size_t)Bn * P0n * Dm * 4);
    float*  posB  = (float*)alloc((size_t)Bn * P0n * Dm * 4);
    __bf16* hbuf  = (__bf16*)alloc((size_t)Bn * NMAX * Dm * 2);
    float*  qkvb  = (float*)alloc((size_t)Bn * NMAX * 3 * Dm * 4);
    __bf16* qbf   = (__bf16*)alloc((size_t)Bn * Hn * NMAXPAD * 16 * 2);
    __bf16* kbf   = (__bf16*)alloc((size_t)Bn * Hn * NMAXPAD * 32 * 2);  // padded K rows
    __bf16* vtf   = (__bf16*)alloc((size_t)Bn * Hn * 16 * NMAXPAD * 2);
    __bf16* obf   = (__bf16*)alloc((size_t)Bn * NMAX * Dm * 2);
    __bf16* ffh   = (__bf16*)alloc((size_t)Bn * NMAX * MLPn * 2);
    float*  human = (float*)alloc((size_t)Bn * P0n * 4);
    int*    idxb  = (int*)alloc((size_t)Bn * P0n * 4);

    // --- weight prep (bf16 + transpose), once per launch ---
    {
        size_t t1 = (size_t)12 * Dm * MLPn;
        size_t t2 = (size_t)12 * Dm * Dm;
        size_t t3 = (size_t)12 * MLPn * Dm;
        wt_prep<<<(unsigned)((t1 + 255) / 256), 256, 0, stream>>>(qkv_w, WqkvT, 12, Dm, 3 * Dm);
        wt_prep<<<(unsigned)((t2 + 255) / 256), 256, 0, stream>>>(out_w, WoutT, 12, Dm, Dm);
        wt_prep<<<(unsigned)((t1 + 255) / 256), 256, 0, stream>>>(ff_w1, Wff1T, 12, Dm, MLPn);
        wt_prep<<<(unsigned)((t3 + 255) / 256), 256, 0, stream>>>(ff_w2, Wff2T, 12, MLPn, Dm);
    }
    hipMemcpyAsync(xA, x_in, (size_t)Bn * NMAX * Dm * 4, hipMemcpyDeviceToDevice, stream);
    init_pos<<<(Bn * P0n * Dm + 255) / 256, 256, 0, stream>>>(pos_in, posA);

    float* x   = xA;  float* x2   = xB;
    float* pos = posA; float* pos2 = posB;
    int n = Jn + P0n;   // 273
    int P = P0n;

    for (int l = 0; l < 12; ++l) {
        if (l > 0)
            add_pos<<<(Bn * P * Dm + 255) / 256, 256, 0, stream>>>(x, pos, la, l, P, n);

        int M = Bn * n;   // always multiple of 16

        // LN1 -> bf16
        ln_rows<<<(M + 7) / 8, 256, 0, stream>>>(x, ln1_w + l * Dm, ln1_b + l * Dm, hbuf, M);

        // QKV GEMM: [M,192] x [192,576]
        {
            dim3 g((M / 16 + 3) / 4, (3 * Dm) / 64);
            gemm_bf16_wmma<0><<<g, 128, 0, stream>>>(hbuf, WqkvT + (size_t)l * 3 * Dm * Dm,
                                                     nullptr, nullptr, qkvb, nullptr,
                                                     M, 3 * Dm, Dm);
        }

        int npad = (n + 15) & ~15;
        rope_pack<<<(Bn * Hn * npad + 255) / 256, 256, 0, stream>>>(qkvb, ang, qbf, kbf, vtf, n, npad);

        bool prune = (l == 3 || l == 6 || l == 9);
        if (prune)
            zero_f<<<(Bn * P0n + 255) / 256, 256, 0, stream>>>(human, Bn * P0n);

        // Attention
        {
            dim3 g(npad / 16, Hn, Bn);
            attn_kernel<<<g, 32, 0, stream>>>(qbf, kbf, vtf, obf, human, n, npad, prune ? 1 : 0);
        }

        // Output projection + bias + residual (in place on x)
        {
            dim3 g((M / 16 + 3) / 4, Dm / 64);
            gemm_bf16_wmma<1><<<g, 128, 0, stream>>>(obf, WoutT + (size_t)l * Dm * Dm,
                                                     out_b + l * Dm, x, x, nullptr,
                                                     M, Dm, Dm);
        }

        if (prune) {
            int Ksel = (P * 7 + 9) / 10;   // ceil(0.7 * P)
            topk_kernel<<<Bn, 256, 0, stream>>>(human, idxb, P, Ksel);
            int nn = Jn + Ksel;
            gather_kernel<<<(Bn * nn * Dm + 255) / 256, 256, 0, stream>>>(x, pos, idxb, x2, pos2,
                                                                          n, P, Ksel);
            { float* t = x; x = x2; x2 = t; t = pos; pos = pos2; pos2 = t; }
            n = nn; P = Ksel; M = Bn * n;
        }

        // LN2 -> bf16
        ln_rows<<<(M + 7) / 8, 256, 0, stream>>>(x, ln2_w + l * Dm, ln2_b + l * Dm, hbuf, M);

        // FF1 + bias + GELU -> bf16
        {
            dim3 g((M / 16 + 3) / 4, MLPn / 64);
            gemm_bf16_wmma<2><<<g, 128, 0, stream>>>(hbuf, Wff1T + (size_t)l * MLPn * Dm,
                                                     ff_b1 + l * MLPn, nullptr, nullptr, ffh,
                                                     M, MLPn, Dm);
        }
        // FF2 + bias + residual (in place on x)
        {
            dim3 g((M / 16 + 3) / 4, Dm / 64);
            gemm_bf16_wmma<1><<<g, 128, 0, stream>>>(ffh, Wff2T + (size_t)l * Dm * MLPn,
                                                     ff_b2 + l * Dm, x, x, nullptr,
                                                     M, Dm, MLPn);
        }
    }

    hipMemcpyAsync(d_out, x, (size_t)out_size * 4, hipMemcpyDeviceToDevice, stream);
}
